// TransFormer_62654982914070
// MI455X (gfx1250) — matrix operand order repaired and checked by reading, not compiled, not run on hardware
//
#include <hip/hip_runtime.h>
#include <math.h>

typedef __attribute__((ext_vector_type(16))) _Float16 v16h;
typedef __attribute__((ext_vector_type(8)))  float    v8f;

#define kC      256
#define kHeads  8
#define kHd     32
#define kHidden 1024
#define kNWin   2048
#define kNPad   64
#define kMPad   (kNWin * kNPad)   /* 131072 padded window-tokens */
#define kMTok   (32 * 56 * 56)    /* 100352 real tokens          */
#define kScaleQ 0.17677669529663687f  /* 32^-0.5 */
#define kKC     64                /* K-chunk staged per LDS panel */
#define kBPad   72                /* padded LDS row stride (halves) */

#define USE_ASYNC_STAGE 1

union Frag {
  v16h v;
  _Float16 h[16];
  uint4 q[2];
};

__device__ __forceinline__ v8f vzero8() {
  v8f z;
#pragma unroll
  for (int i = 0; i < 8; ++i) z[i] = 0.f;
  return z;
}

__device__ __forceinline__ float wave_sum32(float v) {
#pragma unroll
  for (int m = 16; m >= 1; m >>= 1) v += __shfl_xor(v, m, 32);
  return v;
}
__device__ __forceinline__ float half_max16(float v) {
#pragma unroll
  for (int m = 8; m >= 1; m >>= 1) v = fmaxf(v, __shfl_xor(v, m, 32));
  return v;
}
__device__ __forceinline__ float half_sum16(float v) {
#pragma unroll
  for (int m = 8; m >= 1; m >>= 1) v += __shfl_xor(v, m, 32);
  return v;
}

__device__ __forceinline__ void wait_async0() {
#if defined(__has_builtin) && __has_builtin(__builtin_amdgcn_s_wait_asynccnt)
  __builtin_amdgcn_s_wait_asynccnt(0);
#else
  asm volatile("s_wait_asynccnt 0" ::: "memory");
#endif
}

/* ---------------- f32 -> f16 weight convert ---------------- */
__global__ void __launch_bounds__(256) cvt_f32_f16(const float* __restrict__ s,
                                                   _Float16* __restrict__ d,
                                                   int n) {
  int i = blockIdx.x * 256 + threadIdx.x;
  if (i < n) d[i] = (_Float16)s[i];
}

/* ------- LN1 + window partition (pad 49 -> 64 rows with zeros) ------- */
__global__ void __launch_bounds__(256) ln1_window_f16(
    const float* __restrict__ x, const float* __restrict__ g,
    const float* __restrict__ bb, _Float16* __restrict__ yp) {
  const int wave = threadIdx.x >> 5;
  const int lane = threadIdx.x & 31;
  const int win  = blockIdx.x;
  const int bi   = win >> 6;
  const int wrem = win & 63;
  const int wi   = wrem >> 3;
  const int wj   = wrem & 7;
  const int c0   = lane * 8;

  float gv[8], bv[8];
#pragma unroll
  for (int i = 0; i < 8; ++i) { gv[i] = g[c0 + i]; bv[i] = bb[c0 + i]; }

  for (int it = 0; it < 8; ++it) {
    const int t = it * 8 + wave;
    _Float16* dst = yp + ((size_t)win * kNPad + t) * kC + c0;
    if (t >= 49) {
#pragma unroll
      for (int i = 0; i < 8; ++i) dst[i] = (_Float16)0.f;
      continue;
    }
    const int ty = t / 7, tx = t % 7;
    const int hh = wi * 7 + ty, ww = wj * 7 + tx;
    const float* src = x + (((size_t)bi * 56 + hh) * 56 + ww) * kC + c0;
    float v[8];
    float s = 0.f;
#pragma unroll
    for (int i = 0; i < 8; ++i) { v[i] = src[i]; s += v[i]; }
    s = wave_sum32(s) * (1.f / 256.f);
    float var = 0.f;
#pragma unroll
    for (int i = 0; i < 8; ++i) { float d = v[i] - s; var += d * d; }
    var = wave_sum32(var) * (1.f / 256.f);
    const float inv = rsqrtf(var + 1e-5f);
#pragma unroll
    for (int i = 0; i < 8; ++i)
      dst[i] = (_Float16)((v[i] - s) * inv * gv[i] + bv[i]);
  }
}

/* ---------------- LN2 over token-major f32 -> f16 ---------------- */
__global__ void __launch_bounds__(256) ln2_f16(
    const float* __restrict__ x1, const float* __restrict__ g,
    const float* __restrict__ bb, _Float16* __restrict__ z) {
  const int wave = threadIdx.x >> 5;
  const int lane = threadIdx.x & 31;
  const size_t tok = (size_t)blockIdx.x * 8 + wave;
  const int c0 = lane * 8;
  const float* src = x1 + tok * kC + c0;
  _Float16* dst = z + tok * kC + c0;
  float v[8];
  float s = 0.f;
#pragma unroll
  for (int i = 0; i < 8; ++i) { v[i] = src[i]; s += v[i]; }
  s = wave_sum32(s) * (1.f / 256.f);
  float var = 0.f;
#pragma unroll
  for (int i = 0; i < 8; ++i) { float d = v[i] - s; var += d * d; }
  var = wave_sum32(var) * (1.f / 256.f);
  const float inv = rsqrtf(var + 1e-5f);
#pragma unroll
  for (int i = 0; i < 8; ++i)
    dst[i] = (_Float16)((v[i] - s) * inv * g[c0 + i] + bb[c0 + i]);
}

/* ------------- WMMA GEMM: out = A(MxK) * W(NxK)^T -------------
   Block: 256 thr / 8 waves; block tile 256x64; wave tile 32x64.
   Weight panel (64 cols x 64 k, f16) double-buffered in LDS, filled
   with async global->LDS copies overlapped with WMMA on prior panel.
   All 12 fragment loads of a panel are hoisted ahead of the 16 WMMAs
   so the matrix pipe issues one unbroken burst per panel.
   modes: 0=QKV(+bias, scale Q cols)  1=proj(+bias, window-reverse+resid)
          2=FFN1(+bias, exact GELU)   3=FFN2(+bias, resid -> f32 out)     */
__global__ void __launch_bounds__(256) gemm_wmma_f16(
    const _Float16* __restrict__ A, const _Float16* __restrict__ Bw,
    const float* __restrict__ bias, int K, int ldc, int mode,
    _Float16* __restrict__ outh, float* __restrict__ outf,
    const float* __restrict__ resid) {
  __shared__ __align__(16) _Float16 bpanel[2][64][kBPad];

  const int tid  = threadIdx.x;
  const int wave = tid >> 5;
  const int lane = tid & 31;
  const int l15  = lane & 15;
  const int lh   = lane >> 4;            /* lane half-group           */
  const int n0   = blockIdx.y * 64;      /* block column group        */
  const int mt0  = blockIdx.x * 16 + wave * 2; /* first of 2 M-tiles  */
  const int nc   = K / kKC;              /* LDS panels along K        */

  /* per-thread staging coords: 64 rows x 64 halves / 256 thr */
  const int srow = tid >> 2;
  const int skof = (tid & 3) * 16;

  v8f acc0[4], acc1[4];
#pragma unroll
  for (int j = 0; j < 4; ++j) { acc0[j] = vzero8(); acc1[j] = vzero8(); }

  const _Float16* arow0 = A + (size_t)(mt0 * 16 + l15) * K + lh * 8;
  const _Float16* arow1 = arow0 + (size_t)16 * K;

  /* ---- panel staging (async global->LDS, or reg fallback) ---- */
  auto stage = [&](int buf, int c) {
#if USE_ASYNC_STAGE
    const unsigned gof =
        (unsigned)(((n0 + srow) * K + c * kKC + skof) * sizeof(_Float16));
    const unsigned ld0 = (unsigned)(uintptr_t)&bpanel[buf][srow][skof];
    const unsigned ld1 = (unsigned)(uintptr_t)&bpanel[buf][srow][skof + 8];
    asm volatile("global_load_async_to_lds_b128 %0, %2, %3 offset:0\n\t"
                 "global_load_async_to_lds_b128 %1, %2, %3 offset:16"
                 :
                 : "v"(ld0), "v"(ld1), "v"(gof), "s"(Bw)
                 : "memory");
#else
    const _Float16* src = Bw + (size_t)(n0 + srow) * K + c * kKC + skof;
    uint4 t0 = *(const uint4*)src;
    uint4 t1 = *(const uint4*)(src + 8);
    *(uint4*)&bpanel[buf][srow][skof]     = t0;
    *(uint4*)&bpanel[buf][srow][skof + 8] = t1;
#endif
  };

  stage(0, 0);

  for (int c = 0; c < nc; ++c) {
    const int cur = c & 1;
#if USE_ASYNC_STAGE
    wait_async0();
#endif
    __syncthreads();                       /* panel c visible to all   */
    if (c + 1 < nc) stage(cur ^ 1, c + 1); /* overlap fill with math   */

    __builtin_prefetch(arow0 + (c + 1) * kKC, 0, 0);
    __builtin_prefetch(arow1 + (c + 1) * kKC, 0, 0);

    /* hoist ALL fragment loads for this panel, then burst 16 WMMAs */
    Frag a0[2], a1[2], bf[2][4];
    const int kg0 = c * kKC;
#pragma unroll
    for (int h = 0; h < 2; ++h) {
      a0[h].q[0] = *(const uint4*)(arow0 + kg0 + h * 32);
      a0[h].q[1] = *(const uint4*)(arow0 + kg0 + h * 32 + 16);
      a1[h].q[0] = *(const uint4*)(arow1 + kg0 + h * 32);
      a1[h].q[1] = *(const uint4*)(arow1 + kg0 + h * 32 + 16);
    }
#pragma unroll
    for (int h = 0; h < 2; ++h)
#pragma unroll
      for (int j = 0; j < 4; ++j) {
        const _Float16* bp = &bpanel[cur][j * 16 + l15][h * 32 + lh * 8];
        bf[h][j].q[0] = *(const uint4*)bp;
        bf[h][j].q[1] = *(const uint4*)(bp + 16);
      }
#pragma unroll
    for (int h = 0; h < 2; ++h)
#pragma unroll
      for (int j = 0; j < 4; ++j) {
        acc0[j] = __builtin_amdgcn_wmma_f32_16x16x32_f16(
            false, a0[h].v, false, bf[h][j].v, (short)0, acc0[j], false, false);
        acc1[j] = __builtin_amdgcn_wmma_f32_16x16x32_f16(
            false, a1[h].v, false, bf[h][j].v, (short)0, acc1[j], false, false);
      }
  }

  /* ---- epilogue ---- */
#pragma unroll
  for (int mi = 0; mi < 2; ++mi) {
#pragma unroll
    for (int j = 0; j < 4; ++j) {
      const int n = n0 + j * 16 + l15;
      const float bv = bias[n];
#pragma unroll
      for (int r = 0; r < 8; ++r) {
        const int m = (mt0 + mi) * 16 + r + lh * 8;
        float v = (mi == 0 ? acc0[j][r] : acc1[j][r]) + bv;
        if (mode == 0) {
          if (n < 256) v *= kScaleQ; /* fold q scaling */
          outh[(size_t)m * ldc + n] = (_Float16)v;
        } else if (mode == 1) {
          const int t = m & 63;
          if (t < 49) { /* drop pad rows */
            const int win = m >> 6;
            const int bi = win >> 6, wrem = win & 63;
            const int hh = (wrem >> 3) * 7 + t / 7;
            const int ww = (wrem & 7) * 7 + t % 7;
            const size_t tok = ((size_t)bi * 56 + hh) * 56 + ww;
            outf[tok * kC + n] = resid[tok * kC + n] + v;
          }
        } else if (mode == 2) {
          const float gel = 0.5f * v * (1.f + erff(v * 0.70710678118654752f));
          outh[(size_t)m * ldc + n] = (_Float16)gel;
        } else {
          outf[(size_t)m * ldc + n] = resid[(size_t)m * ldc + n] + v;
        }
      }
    }
  }
}

/* ---------- windowed attention: one wave = one (window, head) ---------- */
__global__ void __launch_bounds__(128) win_attn_f16(
    const _Float16* __restrict__ qkv, const float* __restrict__ btab,
    _Float16* __restrict__ o) {
  __shared__ __align__(16) _Float16 plds[4][kNPad * kNPad];
  const int wave = threadIdx.x >> 5;
  const int lane = threadIdx.x & 31;
  const int l15  = lane & 15;
  const int lh   = lane >> 4;
  const int win  = blockIdx.x;
  const int head = blockIdx.y * 4 + wave;
  const _Float16* base = qkv + (size_t)win * kNPad * (3 * kC);

  /* Q fragments (A: 16x32) and K^T fragments (B: 32x16) from global */
  Frag qa[4], kb[4];
#pragma unroll
  for (int mt = 0; mt < 4; ++mt) {
    const _Float16* p =
        base + (size_t)(mt * 16 + l15) * (3 * kC) + head * kHd + lh * 8;
    qa[mt].q[0] = *(const uint4*)p;
    qa[mt].q[1] = *(const uint4*)(p + 16);
  }
#pragma unroll
  for (int nt = 0; nt < 4; ++nt) {
    const _Float16* p =
        base + (size_t)(nt * 16 + l15) * (3 * kC) + kC + head * kHd + lh * 8;
    kb[nt].q[0] = *(const uint4*)p;
    kb[nt].q[1] = *(const uint4*)(p + 16);
  }

  v8f S[4][4];
#pragma unroll
  for (int mt = 0; mt < 4; ++mt)
#pragma unroll
    for (int nt = 0; nt < 4; ++nt) S[mt][nt] = vzero8();

#pragma unroll
  for (int mt = 0; mt < 4; ++mt)
#pragma unroll
    for (int nt = 0; nt < 4; ++nt)
      S[mt][nt] = __builtin_amdgcn_wmma_f32_16x16x32_f16(
          false, qa[mt].v, false, kb[nt].v, (short)0, S[mt][nt], false, false);

  /* relative bias + pad-key mask + row softmax (row = VGPR r + lane-half) */
#pragma unroll
  for (int mt = 0; mt < 4; ++mt) {
#pragma unroll
    for (int r = 0; r < 8; ++r) {
      const int i = mt * 16 + r + lh * 8;
      const int yi = i / 7, xi = i % 7;
      float sv[4];
#pragma unroll
      for (int nt = 0; nt < 4; ++nt) {
        const int j = nt * 16 + l15;
        float s = S[mt][nt][r];
        if (j < 49) {
          if (i < 49) {
            const int idx = (yi - j / 7 + 6) * 13 + (xi - j % 7 + 6);
            s += btab[idx * kHeads + head];
          }
        } else {
          s = -1e30f; /* mask padded keys */
        }
        sv[nt] = s;
      }
      float mx = fmaxf(fmaxf(sv[0], sv[1]), fmaxf(sv[2], sv[3]));
      mx = half_max16(mx);
      float sum = 0.f;
#pragma unroll
      for (int nt = 0; nt < 4; ++nt) {
        sv[nt] = __expf(sv[nt] - mx);
        sum += sv[nt];
      }
      sum = half_sum16(sum);
      const float inv = 1.f / sum;
#pragma unroll
      for (int nt = 0; nt < 4; ++nt)
        plds[wave][i * kNPad + nt * 16 + l15] = (_Float16)(sv[nt] * inv);
    }
  }
  __syncthreads();

  /* O = P(64x64) @ V(64x32): P A-fragments from LDS, V B-fragments gather */
  v8f O[4][2];
#pragma unroll
  for (int mt = 0; mt < 4; ++mt)
#pragma unroll
    for (int nt = 0; nt < 2; ++nt) O[mt][nt] = vzero8();

#pragma unroll
  for (int kc = 0; kc < kNPad; kc += 32) {
    Frag pa[4], vb[2];
#pragma unroll
    for (int mt = 0; mt < 4; ++mt) {
      const _Float16* p = &plds[wave][(mt * 16 + l15) * kNPad + kc + lh * 8];
      pa[mt].q[0] = *(const uint4*)p;
      pa[mt].q[1] = *(const uint4*)(p + 16);
    }
#pragma unroll
    for (int nt = 0; nt < 2; ++nt) {
      const int d = nt * 16 + l15;
#pragma unroll
      for (int e = 0; e < 16; ++e) {
        const int key = kc + lh * 8 + e + (e >= 8 ? 8 : 0);
        vb[nt].h[e] = base[(size_t)key * (3 * kC) + 2 * kC + head * kHd + d];
      }
    }
#pragma unroll
    for (int mt = 0; mt < 4; ++mt)
#pragma unroll
      for (int nt = 0; nt < 2; ++nt)
        O[mt][nt] = __builtin_amdgcn_wmma_f32_16x16x32_f16(
            false, pa[mt].v, false, vb[nt].v, (short)0, O[mt][nt], false,
            false);
  }

#pragma unroll
  for (int mt = 0; mt < 4; ++mt)
#pragma unroll
    for (int nt = 0; nt < 2; ++nt)
#pragma unroll
      for (int r = 0; r < 8; ++r) {
        const int i = mt * 16 + r + lh * 8;
        if (i < 49)
          o[((size_t)win * kNPad + i) * kC + head * kHd + nt * 16 + l15] =
              (_Float16)O[mt][nt][r];
      }
}

extern "C" void kernel_launch(void* const* d_in, const int* in_sizes, int n_in,
                              void* d_out, int out_size, void* d_ws,
                              size_t ws_size, hipStream_t stream) {
  (void)in_sizes; (void)n_in; (void)out_size; (void)ws_size;
  const float* x      = (const float*)d_in[0];
  const float* n1g    = (const float*)d_in[1];
  const float* n1b    = (const float*)d_in[2];
  const float* qkv_w  = (const float*)d_in[3];
  const float* qkv_b  = (const float*)d_in[4];
  const float* btab   = (const float*)d_in[5];
  const float* proj_w = (const float*)d_in[6];
  const float* proj_b = (const float*)d_in[7];
  const float* n2g    = (const float*)d_in[8];
  const float* n2b    = (const float*)d_in[9];
  const float* ffn_w1 = (const float*)d_in[10];
  const float* ffn_b1 = (const float*)d_in[11];
  const float* ffn_w2 = (const float*)d_in[12];
  const float* ffn_b2 = (const float*)d_in[13];
  float* out = (float*)d_out;

  char* w = (char*)d_ws;
  size_t off = 0;
  auto take = [&](size_t bytes) -> void* {
    void* p = (void*)(w + off);
    off += (bytes + 255) & ~(size_t)255;
    return p;
  };
  _Float16* qkv_wh  = (_Float16*)take((size_t)768 * 256 * 2);
  _Float16* proj_wh = (_Float16*)take((size_t)256 * 256 * 2);
  _Float16* w1h     = (_Float16*)take((size_t)1024 * 256 * 2);
  _Float16* w2h     = (_Float16*)take((size_t)256 * 1024 * 2);
  _Float16* yp      = (_Float16*)take((size_t)kMPad * kC * 2);      /* LN1 out; reused as attn out */
  _Float16* qkvh    = (_Float16*)take((size_t)kMTok * kHidden * 2); /* >= QKV f16; reused as FFN hidden */
  float*    x1      = (float*)take((size_t)kMTok * kC * 4);
  _Float16* z       = (_Float16*)take((size_t)kMTok * kC * 2);
  _Float16* obuf = yp;   /* attention output aliases yp (pad rows stay 0) */
  _Float16* hbuf = qkvh; /* FFN hidden aliases dead QKV buffer            */

  /* weight conversions */
  cvt_f32_f16<<<(768 * 256 + 255) / 256, 256, 0, stream>>>(qkv_w, qkv_wh, 768 * 256);
  cvt_f32_f16<<<(256 * 256 + 255) / 256, 256, 0, stream>>>(proj_w, proj_wh, 256 * 256);
  cvt_f32_f16<<<(1024 * 256 + 255) / 256, 256, 0, stream>>>(ffn_w1, w1h, 1024 * 256);
  cvt_f32_f16<<<(256 * 1024 + 255) / 256, 256, 0, stream>>>(ffn_w2, w2h, 256 * 1024);

  /* LN1 + window partition */
  ln1_window_f16<<<kNWin, 256, 0, stream>>>(x, n1g, n1b, yp);

  /* QKV GEMM: M=131072, K=256, N=768 */
  gemm_wmma_f16<<<dim3(kMPad / 256, 768 / 64), 256, 0, stream>>>(
      yp, qkv_wh, qkv_b, 256, 768, 0, qkvh, nullptr, nullptr);

  /* windowed attention */
  win_attn_f16<<<dim3(kNWin, 2), 128, 0, stream>>>(qkvh, btab, obuf);

  /* proj GEMM + window reverse + residual -> x1 */
  gemm_wmma_f16<<<dim3(kMPad / 256, 256 / 64), 256, 0, stream>>>(
      obuf, proj_wh, proj_b, 256, 256, 1, nullptr, x1, x);

  /* LN2 */
  ln2_f16<<<kMTok / 8, 256, 0, stream>>>(x1, n2g, n2b, z);

  /* FFN1 GEMM + GELU: M=100352, K=256, N=1024 */
  gemm_wmma_f16<<<dim3(kMTok / 256, kHidden / 64), 256, 0, stream>>>(
      z, w1h, ffn_b1, 256, kHidden, 2, hbuf, nullptr, nullptr);

  /* FFN2 GEMM + residual -> d_out: M=100352, K=1024, N=256 */
  gemm_wmma_f16<<<dim3(kMTok / 256, 256 / 64), 256, 0, stream>>>(
      hbuf, w2h, ffn_b2, kHidden, 256, 3, nullptr, out, x1);
}
